// BinConvBnHTanh_57939108823746
// MI455X (gfx1250) — 8-bit matrix operand order repaired and checked by reading, compiled, not run on hardware
//
#include <hip/hip_runtime.h>
#include <hip/hip_bf16.h>

typedef __attribute__((ext_vector_type(8))) int v8i;
typedef int vs4i __attribute__((vector_size(16)));

#define Hc 56
#define Wc 56
#define HWc 3136
#define Cc 256
#define Oc 256
#define Nn 32
#define CHW 802816
#define M_TOTAL 100352
#define KTOT 2304

#if __has_builtin(__builtin_amdgcn_global_load_async_to_lds_b128)
#define HAVE_ASYNC_LDS 1
#endif

#define ASG4(p) ((__attribute__((address_space(1))) vs4i*)(vs4i*)(void*)(p))
#define ASL4(p) ((__attribute__((address_space(3))) vs4i*)(vs4i*)(void*)(p))

static __device__ __forceinline__ void wait_async0() {
#if __has_builtin(__builtin_amdgcn_s_wait_asynccnt)
  __builtin_amdgcn_s_wait_asynccnt(0);
#else
  asm volatile("s_wait_asynccnt 0x0" ::: "memory");
#endif
}

__global__ __launch_bounds__(256) void binx_kernel(const float* __restrict__ x,
                                                   signed char* __restrict__ xb) {
  __shared__ __align__(16) signed char tile[Wc * Cc];
  const int nh = blockIdx.x;
  const int n = nh / Hc, h = nh % Hc;
  const float* src = x + (size_t)n * CHW + (size_t)h * Wc;
  const int t = threadIdx.x;
#pragma unroll 1
  for (int j = 0; j < 56; ++j) {
    int i = j * 256 + t;
    int c = i / Wc, w = i % Wc;
    float v = src[(size_t)c * HWc + w];
    tile[w * Cc + c] = (v >= 0.0f) ? (signed char)1 : (signed char)-1;
  }
  __syncthreads();
  int4* d4 = (int4*)(xb + (size_t)nh * Wc * Cc);
  const int4* s4 = (const int4*)tile;
#pragma unroll 1
  for (int i = t; i < (Wc * Cc) / 16; i += 256) d4[i] = s4[i];
}

__global__ __launch_bounds__(256) void binw_kernel(const float* __restrict__ W,
                                                   signed char* __restrict__ wp) {
  int idx = blockIdx.x * 256 + threadIdx.x;
  int c = idx % Cc;
  int tap = (idx / Cc) % 9;
  int o = idx / KTOT;
  float v = W[((size_t)o * Cc + c) * 9 + tap];
  wp[idx] = (v >= 0.0f) ? (signed char)1 : (signed char)-1;
}

__global__ __launch_bounds__(256) void bconv_kernel(const signed char* __restrict__ xb,
                                                    const signed char* __restrict__ wp,
                                                    int* __restrict__ y) {
  __shared__ __align__(16) signed char lA[2 * 256 * 64];
  __shared__ __align__(16) signed char lB[2 * 64 * 64];

  const int t = threadIdx.x;
  const int m_base = blockIdx.x * 256;
  const int o_base = blockIdx.y * 64;
  const int wave = t >> 5;
  const int lane = t & 31;
  const int lane_lo = lane & 15;
  const int hi = lane >> 4;

  const int am = m_base + t;
  const int an = am / HWc;
  const int ahw = am % HWc;
  const int ah = ahw / Wc;
  const int aw = ahw % Wc;
  const signed char* xbN = xb + (size_t)an * HWc * Cc;

  const int orow = t >> 2;
  const int bq = t & 3;
  const signed char* wpRow = wp + (size_t)(o_base + orow) * KTOT + bq * 16;

  v8i acc[8] = {};

#ifdef HAVE_ASYNC_LDS
  {
    const int hh = ah - 1, ww = aw - 1;
    signed char* la = lA + t * 64;
    if (hh >= 0 && hh < Hc && ww >= 0 && ww < Wc) {
      const signed char* g = xbN + ((size_t)(hh * Wc + ww) * Cc);
#pragma unroll
      for (int q = 0; q < 4; ++q)
        __builtin_amdgcn_global_load_async_to_lds_b128(
            ASG4(g + q * 16), ASL4(la + q * 16), 0, 0);
    } else {
      int4 z = make_int4(0, 0, 0, 0);
      int4* dA = (int4*)la;
      dA[0] = z; dA[1] = z; dA[2] = z; dA[3] = z;
    }
    __builtin_amdgcn_global_load_async_to_lds_b128(
        ASG4(wpRow), ASL4(lB + orow * 64 + bq * 16), 0, 0);
  }
  wait_async0();
  __syncthreads();
#else
  int4 ra0, ra1, ra2, ra3, rb;
  {
    const int hh = ah - 1, ww = aw - 1;
    ra0 = make_int4(0, 0, 0, 0); ra1 = ra0; ra2 = ra0; ra3 = ra0;
    if (hh >= 0 && hh < Hc && ww >= 0 && ww < Wc) {
      const int4* p = (const int4*)(xbN + ((size_t)(hh * Wc + ww) * Cc));
      ra0 = p[0]; ra1 = p[1]; ra2 = p[2]; ra3 = p[3];
    }
    rb = *(const int4*)(wpRow);
    int4* dA = (int4*)(lA + t * 64);
    dA[0] = ra0; dA[1] = ra1; dA[2] = ra2; dA[3] = ra3;
    *(int4*)(lB + orow * 64 + bq * 16) = rb;
  }
  __syncthreads();
#endif

#pragma unroll 1
  for (int kk = 0; kk < 36; ++kk) {
    const int cur = kk & 1;
    const int nxt = cur ^ 1;

    if (kk < 35) {
      const int k1 = kk + 1;
      const int tap = k1 >> 2;
      const int c0 = (k1 & 3) << 6;
      const int dy = tap / 3, dx = tap % 3;
      const int hh = ah + dy - 1, ww = aw + dx - 1;
#ifdef HAVE_ASYNC_LDS
      signed char* la = lA + nxt * (256 * 64) + t * 64;
      if (hh >= 0 && hh < Hc && ww >= 0 && ww < Wc) {
        const signed char* g = xbN + ((size_t)(hh * Wc + ww) * Cc + c0);
#pragma unroll
        for (int q = 0; q < 4; ++q)
          __builtin_amdgcn_global_load_async_to_lds_b128(
              ASG4(g + q * 16), ASL4(la + q * 16), 0, 0);
      } else {
        int4 z = make_int4(0, 0, 0, 0);
        int4* dA = (int4*)la;
        dA[0] = z; dA[1] = z; dA[2] = z; dA[3] = z;
      }
      __builtin_amdgcn_global_load_async_to_lds_b128(
          ASG4(wpRow + tap * Cc + c0),
          ASL4(lB + nxt * (64 * 64) + orow * 64 + bq * 16), 0, 0);
#else
      ra0 = make_int4(0, 0, 0, 0); ra1 = ra0; ra2 = ra0; ra3 = ra0;
      if (hh >= 0 && hh < Hc && ww >= 0 && ww < Wc) {
        const int4* p = (const int4*)(xbN + ((size_t)(hh * Wc + ww) * Cc + c0));
        ra0 = p[0]; ra1 = p[1]; ra2 = p[2]; ra3 = p[3];
      }
      rb = *(const int4*)(wpRow + tap * Cc + c0);
#endif
    }

    const signed char* bufA = lA + cur * (256 * 64);
    const signed char* bufB = lB + cur * (64 * 64);

    v8i aF0, aF1;
#pragma unroll
    for (int v = 0; v < 8; ++v) {
      int k = ((v >> 1) << 4) + ((v & 1) << 2) + (hi << 3);
      aF0[v] = *(const int*)(bufA + ((wave << 5) + lane_lo) * 64 + k);
      aF1[v] = *(const int*)(bufA + ((wave << 5) + 16 + lane_lo) * 64 + k);
    }
#pragma unroll
    for (int s = 0; s < 4; ++s) {
      v8i bF;
#pragma unroll
      for (int v = 0; v < 8; ++v) {
        int k = ((v >> 1) << 4) + ((v & 1) << 2) + (hi << 3);
        bF[v] = *(const int*)(bufB + ((s << 4) + lane_lo) * 64 + k);
      }
      acc[s] = __builtin_amdgcn_wmma_i32_16x16x64_iu8(
          true, aF0, true, bF, acc[s], false, false);
      acc[4 + s] = __builtin_amdgcn_wmma_i32_16x16x64_iu8(
          true, aF1, true, bF, acc[4 + s], false, false);
    }

    if (kk < 35) {
#ifdef HAVE_ASYNC_LDS
      wait_async0();
#else
      int4* dA = (int4*)(lA + nxt * (256 * 64) + t * 64);
      dA[0] = ra0; dA[1] = ra1; dA[2] = ra2; dA[3] = ra3;
      *(int4*)(lB + nxt * (64 * 64) + orow * 64 + bq * 16) = rb;
#endif
    }
    __syncthreads();
  }

#pragma unroll
  for (int sub = 0; sub < 2; ++sub) {
    const int mrow = m_base + (wave << 5) + (sub << 4);
#pragma unroll
    for (int s = 0; s < 4; ++s) {
      const int o = o_base + (s << 4) + lane_lo;
#pragma unroll
      for (int r = 0; r < 8; ++r) {
        int m = mrow + r + (hi << 3);
        int n = m / HWc, hw = m % HWc;
        y[(size_t)n * CHW + (size_t)o * HWc + hw] = acc[sub * 4 + s][r];
      }
    }
  }
}

__global__ __launch_bounds__(256) void bnstats_kernel(const int* __restrict__ y,
                                                      const float* __restrict__ gamma,
                                                      const float* __restrict__ beta,
                                                      float* __restrict__ ss) {
  __shared__ float ssum[256];
  __shared__ float ssq[256];
  const int c = blockIdx.x, t = threadIdx.x;
  float sum = 0.0f, sq = 0.0f;
  for (int n = 0; n < Nn; ++n) {
    const int* p = y + (size_t)n * CHW + (size_t)c * HWc;
    for (int i = t; i < HWc; i += 256) {
      float v = (float)p[i];
      sum += v;
      sq += v * v;
    }
  }
  ssum[t] = sum;
  ssq[t] = sq;
  __syncthreads();
  for (int s = 128; s > 0; s >>= 1) {
    if (t < s) { ssum[t] += ssum[t + s]; ssq[t] += ssq[t + s]; }
    __syncthreads();
  }
  if (t == 0) {
    const float inv = 1.0f / (float)M_TOTAL;
    float mean = ssum[0] * inv;
    float var = ssq[0] * inv - mean * mean;
    float scale = gamma[c] * rsqrtf(var + 1e-5f);
    ss[c] = scale;
    ss[Cc + c] = beta[c] - mean * scale;
  }
}

__global__ __launch_bounds__(256) void finalize_kernel(const int* __restrict__ y,
                                                       const float* __restrict__ ss,
                                                       float* __restrict__ out) {
  size_t idx = (size_t)blockIdx.x * 256 + threadIdx.x;
  int c = (int)((idx / HWc) % Cc);
  float v = (float)y[idx] * ss[c] + ss[Cc + c];
  v = fminf(1.0f, fmaxf(-1.0f, v));
  out[idx] = v;
}

extern "C" void kernel_launch(void* const* d_in, const int* in_sizes, int n_in,
                              void* d_out, int out_size, void* d_ws, size_t ws_size,
                              hipStream_t stream) {
  const float* x = (const float*)d_in[0];
  const float* W = (const float*)d_in[1];
  const float* gamma = (const float*)d_in[2];
  const float* beta = (const float*)d_in[3];
  float* out = (float*)d_out;

  char* ws = (char*)d_ws;
  signed char* xb = (signed char*)(ws);
  signed char* wp = (signed char*)(ws + 25690112);
  int* y = (int*)(ws + 26279936);
  float* ss = (float*)(ws + 129040384);

  binx_kernel<<<Nn * Hc, 256, 0, stream>>>(x, xb);
  binw_kernel<<<2304, 256, 0, stream>>>(W, wp);
  bconv_kernel<<<dim3(M_TOTAL / 256, Oc / 64), 256, 0, stream>>>(xb, wp, y);
  bnstats_kernel<<<Cc, 256, 0, stream>>>(y, gamma, beta, ss);
  finalize_kernel<<<M_TOTAL, 256, 0, stream>>>(y, ss, out);
}
